// emiLSTM_ver1_25864293056633
// MI455X (gfx1250) — matrix operand order repaired and checked by reading, not compiled, not run on hardware
//
#include <hip/hip_runtime.h>
#include <hip/hip_bf16.h>

// ---------------- problem constants ----------------
#define D_IN   256
#define HID    512
#define BATCH  64
#define SEQ    512
#define G4     2048            // 4*HID (f,i,c,o)
#define KDIM   768             // HID + D_IN  (A = [h, x])
#define KT     (KDIM/32)       // 24 k-steps of 32
#define NT     (G4/16)         // 128 column tiles
#define NBLK   32              // persistent blocks; block j owns h cols [16j,16j+16)
#define THREADS 256

typedef __attribute__((ext_vector_type(16))) __bf16 v16bf;
typedef __attribute__((ext_vector_type(8)))  float  v8f;
typedef unsigned short bf16_t;

union Frag  { uint4 q[2]; v16bf v; };

// round-to-nearest-even f32 -> bf16 (bit level)
__device__ inline bf16_t f2bf(float x) {
    union { float f; unsigned u; } v; v.f = x;
    unsigned r = v.u + 0x7FFFu + ((v.u >> 16) & 1u);
    return (bf16_t)(r >> 16);
}

// ---------------- prep kernels ----------------
__global__ void cvt_x_bf16(const float* __restrict__ X, bf16_t* __restrict__ Xb, int n) {
    int i = blockIdx.x * blockDim.x + threadIdx.x;
    if (i < n) Xb[i] = f2bf(X[i]);
}

// Pre-swizzle W^T (KDIM x G4) into WMMA B-operand lane order.
// B frag (kt,nt): lane n(0..15) col=n holds K=2v,2v+1 in dword v; lane 16+n holds K=16+2v,17+2v.
// Storage: dword index = ((kt*NT + nt)*32 + lane)*8 + v  -> each lane reads 32 contiguous bytes.
__global__ void pack_w_swz(const float* __restrict__ Wf, const float* __restrict__ Wi,
                           const float* __restrict__ Wc, const float* __restrict__ Wo,
                           unsigned* __restrict__ Wswz, int n) {
    int idx = blockIdx.x * blockDim.x + threadIdx.x;
    if (idx >= n) return;
    unsigned v    = idx & 7u;
    unsigned lane = (idx >> 3) & 31u;
    unsigned ntkt = (unsigned)idx >> 8;
    unsigned nt   = ntkt & 127u;
    unsigned kt   = ntkt >> 7;
    unsigned col  = lane & 15u;
    unsigned g    = nt * 16u + col;
    unsigned krel = ((lane >> 4) << 4) + 2u * v;
    unsigned k    = kt * 32u + krel;
    unsigned gate = g >> 9;
    unsigned r    = g & 511u;
    const float* Wg  = (gate == 0) ? Wf : (gate == 1) ? Wi : (gate == 2) ? Wc : Wo;
    const float* row = Wg + (size_t)r * (HID + D_IN);
    float w0 = (k      < HID) ? row[D_IN + k]     : row[k - HID];
    float w1 = (k + 1u < HID) ? row[D_IN + k + 1] : row[k + 1 - HID];
    Wswz[idx] = (unsigned)f2bf(w0) | ((unsigned)f2bf(w1) << 16);
}

// ---------------- persistent recurrent kernel ----------------
__device__ inline void grid_barrier(unsigned* cnt, unsigned* gen, unsigned target) {
    __builtin_amdgcn_fence(__ATOMIC_RELEASE, "agent");   // flush our h stores to L2
    __syncthreads();
    if (threadIdx.x == 0) {
        unsigned prev = __hip_atomic_fetch_add(cnt, 1u, __ATOMIC_RELAXED, __HIP_MEMORY_SCOPE_AGENT);
        if (prev == target * NBLK - 1u) {
            __hip_atomic_store(gen, target, __ATOMIC_RELEASE, __HIP_MEMORY_SCOPE_AGENT);
        } else {
            while (__hip_atomic_load(gen, __ATOMIC_ACQUIRE, __HIP_MEMORY_SCOPE_AGENT) < target)
                __builtin_amdgcn_s_sleep(1);
        }
    }
    __syncthreads();
    __builtin_amdgcn_fence(__ATOMIC_ACQUIRE, "agent");   // invalidate so fresh h is visible
}

// LDS: block-local weight slice (4 gate-tiles x 24 k-steps, WMMA lane order) + gate staging
#define WUNITS (KT * 4 * 32)            // (kt,gate,lane) units of 8 dwords
__global__ __launch_bounds__(THREADS, 1)
void lstm_persistent(const bf16_t* __restrict__ Xb,          // [B][T][D] bf16
                     const unsigned* __restrict__ Wswz,      // swizzled W^T bf16 pairs
                     const float* __restrict__ bfp, const float* __restrict__ bip,
                     const float* __restrict__ bcp, const float* __restrict__ bop,
                     bf16_t* __restrict__ h0buf, bf16_t* __restrict__ h1buf, // [B][H] bf16
                     float* __restrict__ out,                 // [B][T][H]
                     float* __restrict__ outHT,               // [H][B]
                     float* __restrict__ outST,               // [H][B]
                     unsigned* __restrict__ cnt, unsigned* __restrict__ gen) {
    __shared__ float    lds_g[4 * BATCH * 16];   // 16 KB gate staging [gate][m][col]
    __shared__ unsigned lds_w[WUNITS * 8];       // 96 KB weights: ((kt*4+g)*32+lane)*8 + v

    const int blk   = blockIdx.x;        // owns h columns [blk*16, blk*16+16)
    const int tid   = threadIdx.x;
    const int lane  = tid & 31;
    const int wave  = tid >> 5;          // 8 waves
    const int mtile = wave & 3;          // batch rows mtile*16 .. +15
    const int gpair = wave >> 2;         // gates {0,1} (f,i) or {2,3} (c,o)

    // ---- stage this block's weight slice into LDS (once) ----
    // uint4 granularity: unit = (kt*4+g)*32+lane, two uint4 per unit
    for (int i = tid; i < WUNITS * 2; i += THREADS) {
        int unit  = i >> 1;
        int half  = (i & 1) << 2;                 // dword 0 or 4
        int lane_ = unit & 31;
        int g     = (unit >> 5) & 3;
        int kt    = unit >> 7;
        const uint4* src = (const uint4*)(Wswz +
            (((size_t)kt * NT + (g * 32 + blk)) * 32 + lane_) * 8 + half);
        ((uint4*)lds_w)[i] = *src;
    }
    __syncthreads();

    const int colA  = lane & 15;
    const int khalf = (lane >> 4) << 3;  // 0 or 8: C/D row offset; A lane-half K offset is 2*khalf

    // bias per output column (accumulator init)
    const float* bp0 = (gpair == 0) ? bfp : bcp;
    const float* bp1 = (gpair == 0) ? bip : bop;
    const float bias0 = bp0[blk * 16 + colA];
    const float bias1 = bp1[blk * 16 + colA];

    const int g0 = 2 * gpair;            // gate indices for this wave's two tiles
    const int g1 = 2 * gpair + 1;

    const int arow  = mtile * 16 + colA;          // batch row for A loads
    const int mbase = mtile * 16 + khalf;         // C/D row base for this lane

    float s_reg[4] = {0.f, 0.f, 0.f, 0.f};        // persistent cell state (4 elems/thread)

    for (int t = 0; t < SEQ; ++t) {
        const bf16_t* hread  = (t & 1) ? h1buf : h0buf;
        bf16_t*       hwrite = (t & 1) ? h0buf : h1buf;

        // prefetch next timestep's x fragment (global_prefetch_b8; speculative, safe)
        if (t + 1 < SEQ)
            __builtin_prefetch(Xb + ((size_t)arow * SEQ + (t + 1)) * D_IN + khalf, 0, 3);

        v8f acc0, acc1;
        #pragma unroll
        for (int i = 0; i < 8; ++i) { acc0[i] = bias0; acc1[i] = bias1; }

        // ---- K part 1: h (K = 0..511), B frags from LDS ----
        #pragma unroll 4
        for (int kt = 0; kt < 16; ++kt) {
            Frag a, b0, b1;
            const bf16_t* asrc = hread + arow * HID + kt * 32 + 2 * khalf;
            a.q[0] = *(const uint4*)(asrc);
            a.q[1] = *(const uint4*)(asrc + 8);
            const uint4* bw0 = (const uint4*)(lds_w + (((kt * 4) + g0) * 32 + lane) * 8);
            const uint4* bw1 = (const uint4*)(lds_w + (((kt * 4) + g1) * 32 + lane) * 8);
            b0.q[0] = bw0[0]; b0.q[1] = bw0[1];
            b1.q[0] = bw1[0]; b1.q[1] = bw1[1];
            acc0 = __builtin_amdgcn_wmma_f32_16x16x32_bf16(false, a.v, false, b0.v, (short)0, acc0, false, false);
            acc1 = __builtin_amdgcn_wmma_f32_16x16x32_bf16(false, a.v, false, b1.v, (short)0, acc1, false, false);
        }
        // ---- K part 2: x_t (K = 512..767), B frags from LDS ----
        #pragma unroll
        for (int kt = 16; kt < KT; ++kt) {
            Frag a, b0, b1;
            const bf16_t* asrc = Xb + ((size_t)arow * SEQ + t) * D_IN + (kt - 16) * 32 + 2 * khalf;
            a.q[0] = *(const uint4*)(asrc);
            a.q[1] = *(const uint4*)(asrc + 8);
            const uint4* bw0 = (const uint4*)(lds_w + (((kt * 4) + g0) * 32 + lane) * 8);
            const uint4* bw1 = (const uint4*)(lds_w + (((kt * 4) + g1) * 32 + lane) * 8);
            b0.q[0] = bw0[0]; b0.q[1] = bw0[1];
            b1.q[0] = bw1[0]; b1.q[1] = bw1[1];
            acc0 = __builtin_amdgcn_wmma_f32_16x16x32_bf16(false, a.v, false, b0.v, (short)0, acc0, false, false);
            acc1 = __builtin_amdgcn_wmma_f32_16x16x32_bf16(false, a.v, false, b1.v, (short)0, acc1, false, false);
        }

        // stage gate pre-activations to LDS (C/D layout: dword r -> row mbase+r, col = lane&15)
        #pragma unroll
        for (int r = 0; r < 8; ++r) {
            lds_g[g0 * 1024 + (mbase + r) * 16 + colA] = acc0[r];
            lds_g[g1 * 1024 + (mbase + r) * 16 + colA] = acc1[r];
        }
        __syncthreads();

        // elementwise gate/state update: 1024 elems, 4 per thread (fixed mapping -> s in regs)
        #pragma unroll
        for (int k4 = 0; k4 < 4; ++k4) {
            int idx = tid + 256 * k4;           // = m*16 + c
            int m   = idx >> 4;
            int c   = idx & 15;
            float fg = 1.f / (1.f + __expf(-lds_g[idx]));
            float ig = 1.f / (1.f + __expf(-lds_g[1024 + idx]));
            float cg = tanhf(lds_g[2048 + idx]);
            float og = 1.f / (1.f + __expf(-lds_g[3072 + idx]));
            float s  = fg * s_reg[k4] + ig * cg;
            s_reg[k4] = s;
            float h  = og * tanhf(s);
            int hcol = blk * 16 + c;
            // streaming output: nontemporal, never re-read by this kernel
            __builtin_nontemporal_store(h, &out[(size_t)m * (SEQ * HID) + (size_t)t * HID + hcol]);
            hwrite[m * HID + hcol] = f2bf(h);
            if (t == SEQ - 1) {
                outHT[(size_t)hcol * BATCH + m] = h;   // hT.T  (H,B)
                outST[(size_t)hcol * BATCH + m] = s;   // sT.T  (H,B)
            }
        }

        grid_barrier(cnt, gen, (unsigned)(t + 1));     // also guards LDS reuse
    }
}

// ---------------- host launcher ----------------
extern "C" void kernel_launch(void* const* d_in, const int* in_sizes, int n_in,
                              void* d_out, int out_size, void* d_ws, size_t ws_size,
                              hipStream_t stream) {
    const float* X  = (const float*)d_in[0];
    const float* Wf = (const float*)d_in[1];
    const float* bf = (const float*)d_in[2];
    const float* Wi = (const float*)d_in[3];
    const float* bi = (const float*)d_in[4];
    const float* Wc = (const float*)d_in[5];
    const float* bc = (const float*)d_in[6];
    const float* Wo = (const float*)d_in[7];
    const float* bo = (const float*)d_in[8];

    char*  ws  = (char*)d_ws;
    size_t off = 0;
    bf16_t*   Xb   = (bf16_t*)(ws + off);   off += (size_t)BATCH * SEQ * D_IN * 2; // 16 MB
    unsigned* Wswz = (unsigned*)(ws + off); off += (size_t)KT * NT * 32 * 8 * 4;   // 3 MB
    bf16_t*   h0   = (bf16_t*)(ws + off);   off += (size_t)BATCH * HID * 2;
    bf16_t*   h1   = (bf16_t*)(ws + off);   off += (size_t)BATCH * HID * 2;
    unsigned* cnt  = (unsigned*)(ws + off); off += 128;
    unsigned* gen  = (unsigned*)(ws + off); off += 128;

    // deterministic init each launch (async memset is graph-capturable)
    hipMemsetAsync(h0, 0, (size_t)BATCH * HID * 2, stream);   // h_0 = 0 (t=0 reads h0 buffer)
    hipMemsetAsync(cnt, 0, 256, stream);                      // barrier counter + generation

    float* out   = (float*)d_out;
    float* outHT = out + (size_t)BATCH * SEQ * HID;
    float* outST = outHT + (size_t)HID * BATCH;

    const int nX = BATCH * SEQ * D_IN;                        // 8,388,608
    cvt_x_bf16<<<(nX + THREADS - 1) / THREADS, THREADS, 0, stream>>>(X, Xb, nX);

    const int nW = KT * NT * 32 * 8;                          // 786,432 packed dwords
    pack_w_swz<<<(nW + THREADS - 1) / THREADS, THREADS, 0, stream>>>(Wf, Wi, Wc, Wo, Wswz, nW);

    lstm_persistent<<<NBLK, THREADS, 0, stream>>>(Xb, Wswz, bf, bi, bc, bo,
                                                  h0, h1, out, outHT, outST, cnt, gen);
}
